// AttentionLSTM_15547781612006
// MI455X (gfx1250) — matrix operand order repaired and checked by reading, not compiled, not run on hardware
//
#include <hip/hip_runtime.h>
#include <hip/hip_bf16.h>

#define D_DIM   512
#define H_DIM   512
#define N_BATCH 256
#define T_STEPS 128
#define FOURH   2048

typedef __attribute__((ext_vector_type(16))) __bf16       v16bf;
typedef __attribute__((ext_vector_type(8)))  float        v8f;
typedef __attribute__((ext_vector_type(4)))  unsigned int v4u;

union FragBF {
    v16bf v;
    v4u   q[2];
};

// Load a 16-element bf16 fragment as two 16-byte chunks (g0 -> elements 0..7, g1 -> 8..15)
__device__ __forceinline__ v16bf load_frag_split(const __bf16* g0, const __bf16* g1) {
    FragBF u;
    u.q[0] = *reinterpret_cast<const v4u*>(g0);
    u.q[1] = *reinterpret_cast<const v4u*>(g1);
    return u.v;
}

__device__ __forceinline__ float sigmoidf_(float x) {
    return 1.0f / (1.0f + __expf(-x));
}

// ---------------------------------------------------------------------------
// Prep kernels
// ---------------------------------------------------------------------------

__global__ __launch_bounds__(256) void cvt_bf16_kernel(const float* __restrict__ src,
                                                       __bf16* __restrict__ dst, int n) {
    int i = blockIdx.x * 256 + threadIdx.x;
    if (i < n) dst[i] = (__bf16)src[i];
}

// WT[j*K + k] = W[k*NC + j]   (W is (K, NC) row-major; WT is (NC, K) row-major)
__global__ __launch_bounds__(256) void transpose_cvt_kernel(const float* __restrict__ W,
                                                            __bf16* __restrict__ WT,
                                                            int K, int NC) {
    int i = blockIdx.x * 256 + threadIdx.x;
    if (i >= K * NC) return;
    int j = i / K;
    int k = i - j * K;
    WT[i] = (__bf16)W[(long long)k * NC + j];
}

// h0 = c0 = mean(A over last 16); also bf16 copy of h0
__global__ __launch_bounds__(256) void init_state_kernel(const float* __restrict__ A,
                                                         float* __restrict__ h0,
                                                         float* __restrict__ c,
                                                         __bf16* __restrict__ hbf) {
    int i = blockIdx.x * 256 + threadIdx.x;          // over N*H
    const float* p = A + (long long)i * 16;
    float s = 0.0f;
#pragma unroll
    for (int r = 0; r < 16; ++r) s += p[r];
    s *= (1.0f / 16.0f);
    h0[i] = s;
    c[i]  = s;
    hbf[i] = (__bf16)s;
}

// ---------------------------------------------------------------------------
// Per-step attention: scores = (h . A_flat[:, :, p]) / sqrt(H); softmax over 16;
// attn = A_flat @ w. One block per batch row n. 256 threads, 2 h-elements each.
// ---------------------------------------------------------------------------

__global__ __launch_bounds__(256) void attn_kernel(const float* __restrict__ hprev,
                                                   long long hstride,
                                                   const float* __restrict__ A,
                                                   __bf16* __restrict__ attnbf) {
    __shared__ float part[256][16];
    __shared__ float wsm[16];
    const int n   = blockIdx.x;
    const int tid = threadIdx.x;
    const float* hrow = hprev + (long long)n * hstride;
    const float* Arow = A + (long long)n * H_DIM * 16;

    const float h1 = hrow[tid];
    const float h2 = hrow[tid + 256];
    const float* a1 = Arow + tid * 16;
    const float* a2 = Arow + (tid + 256) * 16;

#pragma unroll
    for (int p = 0; p < 16; ++p) part[tid][p] = h1 * a1[p] + h2 * a2[p];
    __syncthreads();

    for (int s = 128; s > 0; s >>= 1) {
        if (tid < s) {
#pragma unroll
            for (int p = 0; p < 16; ++p) part[tid][p] += part[tid + s][p];
        }
        __syncthreads();
    }

    if (tid == 0) {
        const float scale = 0.04419417382415922f;   // 1/sqrt(512)
        float sc[16], mx = -1e30f;
#pragma unroll
        for (int p = 0; p < 16; ++p) { sc[p] = part[0][p] * scale; mx = fmaxf(mx, sc[p]); }
        float sum = 0.0f;
#pragma unroll
        for (int p = 0; p < 16; ++p) { sc[p] = __expf(sc[p] - mx); sum += sc[p]; }
        float inv = 1.0f / sum;
#pragma unroll
        for (int p = 0; p < 16; ++p) wsm[p] = sc[p] * inv;
    }
    __syncthreads();

    float o1 = 0.0f, o2 = 0.0f;
#pragma unroll
    for (int p = 0; p < 16; ++p) { o1 += a1[p] * wsm[p]; o2 += a2[p] * wsm[p]; }
    attnbf[(long long)n * H_DIM + tid]       = (__bf16)o1;
    attnbf[(long long)n * H_DIM + tid + 256] = (__bf16)o2;
}

// ---------------------------------------------------------------------------
// Fused LSTM step: a = [x_t, h, attn] @ [WxT;WhT;WaT]^T + b, then gating.
// Grid: 512 blocks x 128 threads (4 waves) = 2048 waves.
//   block  -> one 16x16 (batch x hidden) tile:  mt = blockIdx>>5, ht = blockIdx&31
//   wave g -> gate g of that tile (i, f, o, g), K-pipelined WMMA accumulation
// Gates are exchanged through LDS, then 128 threads apply the LSTM cell update.
// ---------------------------------------------------------------------------

__global__ __launch_bounds__(128) void lstm_step_kernel(
    const __bf16* __restrict__ xbf,       // (N, T, D) bf16
    const __bf16* __restrict__ WxT,       // (4H, D) bf16 (transposed)
    const __bf16* __restrict__ WhT,       // (4H, H)
    const __bf16* __restrict__ WaT,       // (4H, H)
    const __bf16* __restrict__ hbf_in,    // (N, H)
    const __bf16* __restrict__ attnbf,    // (N, H)
    const float*  __restrict__ bias,      // (4H)
    float* __restrict__ cstate,           // (N, H) in/out
    float* __restrict__ out,              // (N, T, H)
    __bf16* __restrict__ hbf_out,         // (N, H) next-h bf16 (ping-pong)
    int t) {

    __shared__ float tl[4][16][17];       // 4 gate tiles, padded rows

    const int g    = threadIdx.x >> 5;    // wave in block = gate 0..3
    const int lane = threadIdx.x & 31;
    const int half = lane >> 4;
    const int lq   = lane & 15;
    const int mt   = blockIdx.x >> 5;     // batch tile   0..15
    const int ht   = blockIdx.x & 31;     // hidden tile  0..31

    const int m = mt * 16 + lq;           // batch row for this lane's A fragment
    const __bf16* xrow = xbf    + ((long long)m * T_STEPS + t) * D_DIM;
    const __bf16* hrow = hbf_in + (long long)m * H_DIM;
    const __bf16* arow = attnbf + (long long)m * H_DIM;

    const int hc = ht * 16 + lq;                    // hidden column (lane's B column)
    const long long wro = (long long)(g * H_DIM + hc) * 512;  // B row offset in WT

    const __bf16* const segA[3] = { xrow, hrow, arow };
    const __bf16* const segW[3] = { WxT + wro, WhT + wro, WaT + wro };

    v8f acc = {};

#pragma unroll
    for (int seg = 0; seg < 3; ++seg) {
        const __bf16* Ap = segA[seg];
        const __bf16* Wp = segW[seg];
        // software-pipelined K loop: prefetch kc+1 while WMMA on kc
        int ka = half * 8;                // A: two 8-elem groups at ka, ka+16
        int kb = half * 8;                // B: two 8-elem groups at kb, kb+16
        v16bf a_cur = load_frag_split(Ap + ka, Ap + ka + 16);
        v16bf b_cur = load_frag_split(Wp + kb, Wp + kb + 16);
#pragma unroll
        for (int kc = 0; kc < 16; ++kc) {
            v16bf a_nxt, b_nxt;
            if (kc < 15) {
                const int kan = (kc + 1) * 32 + half * 8;
                const int kbn = (kc + 1) * 32 + half * 8;
                a_nxt = load_frag_split(Ap + kan, Ap + kan + 16);
                b_nxt = load_frag_split(Wp + kbn, Wp + kbn + 16);
            }
            acc = __builtin_amdgcn_wmma_f32_16x16x32_bf16(false, a_cur, false, b_cur,
                                                          (short)0, acc, false, false);
            a_cur = a_nxt;
            b_cur = b_nxt;
        }
    }

    // dump gate tile to LDS (C/D layout: M = r + half*8, N = lq)
#pragma unroll
    for (int r = 0; r < 8; ++r) tl[g][half * 8 + r][lq] = acc[r];
    __syncthreads();

    // 128 threads apply the LSTM cell update to the 256-element tile
#pragma unroll
    for (int e2 = 0; e2 < 2; ++e2) {
        const int e    = threadIdx.x + e2 * 128;
        const int mrow = e >> 4;
        const int ncol = e & 15;
        const int n    = mt * 16 + mrow;
        const int col  = ht * 16 + ncol;

        const float iv = sigmoidf_(tl[0][mrow][ncol] + bias[0 * H_DIM + col]);
        const float fv = sigmoidf_(tl[1][mrow][ncol] + bias[1 * H_DIM + col]);
        const float ov = sigmoidf_(tl[2][mrow][ncol] + bias[2 * H_DIM + col]);
        const float gv = tanhf(tl[3][mrow][ncol] + bias[3 * H_DIM + col]);

        const long long ci = (long long)n * H_DIM + col;
        const float cn = fv * cstate[ci] + iv * gv;
        const float hn = ov * tanhf(cn);
        cstate[ci] = cn;
        out[((long long)n * T_STEPS + t) * (long long)H_DIM + col] = hn;
        hbf_out[ci] = (__bf16)hn;
    }
}

// ---------------------------------------------------------------------------
// Host launch
// ---------------------------------------------------------------------------

extern "C" void kernel_launch(void* const* d_in, const int* in_sizes, int n_in,
                              void* d_out, int out_size, void* d_ws, size_t ws_size,
                              hipStream_t stream) {
    const float* x     = (const float*)d_in[0];   // (N, T, D)
    const float* A     = (const float*)d_in[1];   // (N, H, 4, 4)
    const float* Wx    = (const float*)d_in[2];   // (D, 4H)
    const float* Wh    = (const float*)d_in[3];   // (H, 4H)
    const float* Wattn = (const float*)d_in[4];   // (H, 4H)
    const float* b     = (const float*)d_in[5];   // (4H)
    float* out = (float*)d_out;                   // (N, T, H)

    char* ws = (char*)d_ws;
    size_t off = 0;
    auto alloc = [&](size_t bytes) -> void* {
        void* p = ws + off;
        off += (bytes + 255) & ~(size_t)255;
        return p;
    };

    __bf16* WxT    = (__bf16*)alloc((size_t)FOURH * D_DIM * 2);
    __bf16* WhT    = (__bf16*)alloc((size_t)FOURH * H_DIM * 2);
    __bf16* WaT    = (__bf16*)alloc((size_t)FOURH * H_DIM * 2);
    __bf16* xbf    = (__bf16*)alloc((size_t)N_BATCH * T_STEPS * D_DIM * 2);
    __bf16* hA     = (__bf16*)alloc((size_t)N_BATCH * H_DIM * 2);
    __bf16* hB     = (__bf16*)alloc((size_t)N_BATCH * H_DIM * 2);
    __bf16* attnbf = (__bf16*)alloc((size_t)N_BATCH * H_DIM * 2);
    float*  cbuf   = (float*)alloc((size_t)N_BATCH * H_DIM * 4);
    float*  h0     = (float*)alloc((size_t)N_BATCH * H_DIM * 4);

    // Prep: bf16 x, transposed bf16 weights, initial state
    {
        int n = N_BATCH * T_STEPS * D_DIM;
        cvt_bf16_kernel<<<(n + 255) / 256, 256, 0, stream>>>(x, xbf, n);
    }
    {
        int n = D_DIM * FOURH;
        transpose_cvt_kernel<<<(n + 255) / 256, 256, 0, stream>>>(Wx, WxT, D_DIM, FOURH);
        transpose_cvt_kernel<<<(n + 255) / 256, 256, 0, stream>>>(Wh, WhT, H_DIM, FOURH);
        transpose_cvt_kernel<<<(n + 255) / 256, 256, 0, stream>>>(Wattn, WaT, H_DIM, FOURH);
    }
    init_state_kernel<<<(N_BATCH * H_DIM) / 256, 256, 0, stream>>>(A, h0, cbuf, hA);

    __bf16* hr = hA;
    __bf16* hw = hB;
    for (int t = 0; t < T_STEPS; ++t) {
        const float* hprev = (t == 0) ? h0 : (out + (size_t)(t - 1) * H_DIM);
        long long hstride  = (t == 0) ? (long long)H_DIM : (long long)T_STEPS * H_DIM;
        attn_kernel<<<N_BATCH, 256, 0, stream>>>(hprev, hstride, A, attnbf);
        lstm_step_kernel<<<512, 128, 0, stream>>>(xbf, WxT, WhT, WaT, hr, attnbf,
                                                  b, cbuf, out, hw, t);
        __bf16* tmp = hr; hr = hw; hw = tmp;
    }
}
